// DSGNet_50448685859244
// MI455X (gfx1250) — matrix operand order repaired and checked by reading, not compiled, not run on hardware
//
#include <hip/hip_runtime.h>
#include <cstddef>
#include <cstdint>

// ---------------- model dimensions ----------------
#define N_ENTC   50000
#define NREL2C   1000
#define EDGESC   800000
#define HDIM     128
#define BSZ      1024
#define CCH      96
#define FLATD    9600
#define MLPD     200     // logical MLP width
#define MLPP     224     // padded MLP width (multiple of 32)
#define TOPKC    15
#define NBIN     64

#define CDIV(a,b) (((a)+(b)-1)/(b))

// ---------------- WMMA types ----------------
typedef __attribute__((ext_vector_type(16))) __bf16 v16bf;
typedef __attribute__((ext_vector_type(8)))  float  v8f;
union Frag { v16bf v; unsigned int u[8]; };

__device__ __forceinline__ unsigned short f2bf(float f) {
  unsigned u = __float_as_uint(f);
  unsigned r = (u + 0x7FFFu + ((u >> 16) & 1u)) >> 16;   // RNE truncate f32 -> bf16
  return (unsigned short)r;
}

// order-preserving f32 <-> u32 (for atomicMax segment-max)
__device__ __forceinline__ unsigned fOrd(float f) {
  unsigned u = __float_as_uint(f);
  return (u & 0x80000000u) ? ~u : (u | 0x80000000u);
}
__device__ __forceinline__ float fUnord(unsigned u) {
  u = (u & 0x80000000u) ? (u & 0x7FFFFFFFu) : ~u;
  return __uint_as_float(u);
}

__device__ __forceinline__ float act_apply(float x, int act) {
  if (act == 1) return fmaxf(x, 0.f);
  if (act == 2) return tanhf(x);
  if (act == 3) return 1.f / (1.f + __expf(-x));
  return x;
}

// ---- CDNA5 async global->LDS copy (ASYNCcnt path) ----
// LDS offset = low 32 bits of the flat shared address (ISA 10.2: LDS aperture
// carries the LDS byte offset in addr[31:0]).
__device__ __forceinline__ void async_g2l_b128(void* lds_ptr, const void* gptr) {
  unsigned lds = (unsigned)(uintptr_t)lds_ptr;
  unsigned long long ga = (unsigned long long)(uintptr_t)gptr;
  asm volatile("global_load_async_to_lds_b128 %0, %1, off"
               :: "v"(lds), "v"(ga) : "memory");
}
__device__ __forceinline__ void wait_async0() {
#if __has_builtin(__builtin_amdgcn_s_wait_asynccnt)
  __builtin_amdgcn_s_wait_asynccnt(0);
#else
  asm volatile("s_wait_asynccnt 0x0" ::: "memory");
#endif
}

// ---------------- utility kernels ----------------
__global__ void k_fill_u32(unsigned* __restrict__ p, unsigned v, size_t n) {
  size_t i = (size_t)blockIdx.x * blockDim.x + threadIdx.x;
  if (i < n) p[i] = v;
}

__global__ void k_cvt_bf(const float* __restrict__ in, unsigned short* __restrict__ out, size_t n) {
  size_t i = (size_t)blockIdx.x * blockDim.x + threadIdx.x;
  if (i < n) out[i] = f2bf(in[i]);
}

// zero-padded f32 -> bf16 matrix convert: in[rows][cols] -> out[rowsP][colsP]
__global__ void k_cvt_pad(const float* __restrict__ in, unsigned short* __restrict__ out,
                          int rows, int cols, int rowsP, int colsP) {
  int i = blockIdx.x * blockDim.x + threadIdx.x;
  if (i >= rowsP * colsP) return;
  int r = i / colsP, c = i % colsP;
  out[i] = (r < rows && c < cols) ? f2bf(in[(size_t)r * cols + c]) : (unsigned short)0;
}

__global__ void k_pad_vec(const float* __restrict__ in, float* __restrict__ out, int n, int np) {
  int i = blockIdx.x * blockDim.x + threadIdx.x;
  if (i < np) out[i] = (i < n) ? in[i] : 0.f;
}

__global__ void k_add3(const float* __restrict__ e, const float* __restrict__ c1,
                       const float* __restrict__ c2, float* __restrict__ out,
                       unsigned short* __restrict__ outbf, size_t n) {
  size_t i = (size_t)blockIdx.x * blockDim.x + threadIdx.x;
  if (i < n) { float v = e[i] + c1[i] + c2[i]; out[i] = v; outbf[i] = f2bf(v); }
}

// ---------------- generic bf16 WMMA GEMM ----------------
// C[M,N] = act(A[M,K] @ (BT ? B[N,K]^T : B[K,N]) + bias[N])
// REQUIREMENTS (guaranteed by launcher): K % 32 == 0; A and B are workspace
// buffers with >=64KB tail padding, so unguarded tile reads past M/N are safe
// (garbage only flows into accumulator rows/cols the epilogue never stores).
// block = 256 threads = 8 waves; block tile 64x128; wave tile 32x32 (4 WMMAs).
// A tile and non-transposed B tile are staged with GLOBAL_LOAD_ASYNC_TO_LDS_B128
// (ASYNCcnt); each wave waits its own asynccnt==0 before the barrier publishes.
__global__ __launch_bounds__(256) void k_gemm_bf16(
    const unsigned short* __restrict__ A, const unsigned short* __restrict__ B,
    const float* __restrict__ bias, float* __restrict__ C,
    int M, int N, int K, int BT, int act)
{
  __shared__ unsigned short As[64][40];    // 80B rows: 16B aligned vector slots
  __shared__ unsigned short Bs[32][136];   // 272B rows: 16B aligned vector slots
  const int tid  = threadIdx.x;
  const int w    = tid >> 5;
  const int lane = tid & 31;
  const int m0 = blockIdx.y * 64;
  const int n0 = blockIdx.x * 128;
  const int wr = w >> 2;          // 0..1 -> 32-row slice
  const int wc = w & 3;           // 0..3 -> 32-col slice
  const int h  = lane >> 4;       // lane half
  const int ml = lane & 15;

  v8f acc00 = {0.f,0.f,0.f,0.f,0.f,0.f,0.f,0.f};
  v8f acc01 = acc00, acc10 = acc00, acc11 = acc00;

  // hoisted staging coordinates
  const int ra  = tid >> 2;                 // A tile row (0..63)
  const int ka  = (tid & 3) * 8;            // A tile k   (8 contiguous)
  const size_t aoff = (size_t)(m0 + ra) * K + ka;
  const int rb  = tid >> 3;                 // B tile k row (0..31)   [BT=0]
  const int nb  = (tid & 7) * 16;           // B tile n     (16 contiguous)
  const int nt  = tid & 127;                // B column     (0..127)  [BT=1]
  const int kt  = (tid >> 7) * 16;          // B tile k     (16 contiguous)

  for (int k0 = 0; k0 < K; k0 += 32) {
    // ---- stage A tile 64x32: async b128 copy, no VGPR round-trip ----
    async_g2l_b128(&As[ra][ka], A + aoff + k0);
    // ---- stage B tile 32x128 as [k][n] ----
    if (!BT) {
      const unsigned short* gb = B + (size_t)(k0 + rb) * N + n0 + nb;
      async_g2l_b128(&Bs[rb][nb],     gb);
      async_g2l_b128(&Bs[rb][nb + 8], gb + 8);
    } else {
      const uint4* p = (const uint4*)(B + (size_t)(n0 + nt) * K + k0 + kt);
      uint4 q0 = p[0], q1 = p[1];
      unsigned wv[8] = {q0.x, q0.y, q0.z, q0.w, q1.x, q1.y, q1.z, q1.w};
      #pragma unroll
      for (int i = 0; i < 8; ++i) {
        Bs[kt + 2*i    ][nt] = (unsigned short)(wv[i] & 0xFFFFu);
        Bs[kt + 2*i + 1][nt] = (unsigned short)(wv[i] >> 16);
      }
    }
    wait_async0();                 // this wave's LDS writes are complete
    __syncthreads();               // all waves' tiles published

    // ---- fragments (ISA 7.12.2 layouts) ----
    Frag fa0, fa1, fb0, fb1;
    const int arow = wr * 32 + ml;
    #pragma unroll
    for (int p = 0; p < 8; ++p) {
      int kk = (p < 4 ? 2 * p : 16 + 2 * (p - 4)) + 8 * h;
      fa0.u[p] = *(const unsigned int*)&As[arow][kk];
      fa1.u[p] = *(const unsigned int*)&As[arow + 16][kk];
    }
    {
      const int bcol = wc * 32 + ml;
      #pragma unroll
      for (int j = 0; j < 8; ++j) {
        int kk = (j < 4 ? 2 * j : 16 + 2 * (j - 4)) + 8 * h;
        fb0.u[j] = (unsigned int)Bs[kk][bcol] | ((unsigned int)Bs[kk + 1][bcol] << 16);
        fb1.u[j] = (unsigned int)Bs[kk][bcol + 16] | ((unsigned int)Bs[kk + 1][bcol + 16] << 16);
      }
    }
    acc00 = __builtin_amdgcn_wmma_f32_16x16x32_bf16(false, fa0.v, false, fb0.v,
                                                    (short)0, acc00, false, false);
    acc01 = __builtin_amdgcn_wmma_f32_16x16x32_bf16(false, fa0.v, false, fb1.v,
                                                    (short)0, acc01, false, false);
    acc10 = __builtin_amdgcn_wmma_f32_16x16x32_bf16(false, fa1.v, false, fb0.v,
                                                    (short)0, acc10, false, false);
    acc11 = __builtin_amdgcn_wmma_f32_16x16x32_bf16(false, fa1.v, false, fb1.v,
                                                    (short)0, acc11, false, false);
    __syncthreads();
  }

  // ---- epilogue: C element r -> M = r + 8*half, N = lane%16 ----
  #pragma unroll
  for (int j = 0; j < 2; ++j) {
    const int n = n0 + wc * 32 + j * 16 + ml;
    if (n >= N) continue;
    const float bn = (bias != nullptr) ? bias[n] : 0.f;
    const v8f* a0 = j ? &acc01 : &acc00;
    const v8f* a1 = j ? &acc11 : &acc10;
    #pragma unroll
    for (int r = 0; r < 8; ++r) {
      int ma = m0 + wr * 32 + r + 8 * h;
      int mb = ma + 16;
      if (ma < M) C[(size_t)ma * N + n] = act_apply((*a0)[r] + bn, act);
      if (mb < M) C[(size_t)mb * N + n] = act_apply((*a1)[r] + bn, act);
    }
  }
}

// ---------------- comp_layer edge kernels ----------------
// one wave (32 lanes) per edge; each lane handles a float4 (128 = 32*4)
__global__ __launch_bounds__(256) void k_edge_score(
    const float* __restrict__ node, const float* __restrict__ rel,
    const int* __restrict__ src, const int* __restrict__ dst, const int* __restrict__ rid,
    float* __restrict__ score, unsigned* __restrict__ mnode, int E)
{
  int e = (int)(((size_t)blockIdx.x * blockDim.x + threadIdx.x) >> 5);
  int lane = threadIdx.x & 31;
  if (e >= E) return;
  int s = src[e], d = dst[e], r = rid[e];
  float4 a = ((const float4*)(node + (size_t)s * HDIM))[lane];
  float4 b = ((const float4*)(rel  + (size_t)r * HDIM))[lane];
  float4 c = ((const float4*)(node + (size_t)d * HDIM))[lane];
  float acc = a.x*b.x*c.x + a.y*b.y*c.y + a.z*b.z*c.z + a.w*b.w*c.w;
  #pragma unroll
  for (int off = 16; off; off >>= 1) acc += __shfl_down(acc, off, 32);
  if (lane == 0) {
    score[e] = acc;
    atomicMax(mnode + d, fOrd(acc));
  }
}

__global__ void k_edge_expz(const float* __restrict__ score, const unsigned* __restrict__ mnode,
                            const int* __restrict__ dst, float* __restrict__ z,
                            unsigned* __restrict__ hist, int E)
{
  int e = blockIdx.x * blockDim.x + threadIdx.x;
  if (e >= E) return;
  int d = dst[e];
  float ex = __expf(score[e] - fUnord(mnode[d]));       // in (0, 1]
  atomicAdd(z + d, ex);
  int bin = (int)(ex * (float)NBIN); if (bin > NBIN - 1) bin = NBIN - 1;
  atomicAdd(hist + (size_t)d * NBIN + bin, 1u);
}

// per-node threshold bin + residual budget for top-15 selection
__global__ void k_node_select(const unsigned* __restrict__ hist, int* __restrict__ thr,
                              unsigned* __restrict__ bud, unsigned* __restrict__ cnt, int n)
{
  int i = blockIdx.x * blockDim.x + threadIdx.x;
  if (i >= n) return;
  unsigned cum = 0; int t = -1; unsigned b = 0;
  const unsigned* hrow = hist + (size_t)i * NBIN;
  for (int bin = NBIN - 1; bin >= 0; --bin) {
    unsigned c = hrow[bin];
    if (cum + c > TOPKC) { t = bin; b = TOPKC - cum; break; }
    cum += c;
  }
  thr[i] = t; bud[i] = b; cnt[i] = 0;
}

__global__ __launch_bounds__(256) void k_edge_agg(
    const float* __restrict__ node, const float* __restrict__ rel,
    const int* __restrict__ src, const int* __restrict__ dst, const int* __restrict__ rid,
    const float* __restrict__ score, const unsigned* __restrict__ mnode,
    const float* __restrict__ z, const int* __restrict__ thr,
    const unsigned* __restrict__ bud, unsigned* __restrict__ cnt,
    float* __restrict__ neigh, int E)
{
  int e = (int)(((size_t)blockIdx.x * blockDim.x + threadIdx.x) >> 5);
  int lane = threadIdx.x & 31;
  if (e >= E) return;
  int s = src[e], d = dst[e], r = rid[e];
  int keep = 0; float wgt = 0.f;
  if (lane == 0) {
    float ex = __expf(score[e] - fUnord(mnode[d]));
    wgt = ex / z[d];
    int bin = (int)(ex * (float)NBIN); if (bin > NBIN - 1) bin = NBIN - 1;
    int t = thr[d];
    if (bin > t) keep = 1;
    else if (bin == t) { unsigned slot = atomicAdd(cnt + d, 1u); keep = (slot < bud[d]); }
  }
  keep = __shfl(keep, 0, 32);
  wgt  = __shfl(wgt,  0, 32);
  if (!keep) return;
  float4 a = ((const float4*)(node + (size_t)s * HDIM))[lane];
  float4 b = ((const float4*)(rel  + (size_t)r * HDIM))[lane];
  float* out = neigh + (size_t)d * HDIM + lane * 4;
  atomicAdd(out + 0, a.x * b.x * wgt);
  atomicAdd(out + 1, a.y * b.y * wgt);
  atomicAdd(out + 2, a.z * b.z * wgt);
  atomicAdd(out + 3, a.w * b.w * wgt);
}

// ---------------- mlp tail + correlation ----------------
__global__ void k_rowdot(const float* __restrict__ A, const float* __restrict__ w,
                         const float* __restrict__ b, float* __restrict__ out,
                         int M, int Kd, int lda)
{
  int row = (int)(((size_t)blockIdx.x * blockDim.x + threadIdx.x) >> 5);
  int lane = threadIdx.x & 31;
  if (row >= M) return;
  float acc = 0.f;
  for (int j = lane; j < Kd; j += 32) acc += A[(size_t)row * lda + j] * w[j];
  #pragma unroll
  for (int off = 16; off; off >>= 1) acc += __shfl_down(acc, off, 32);
  if (lane == 0) out[row] = acc + b[0];
}

__global__ void k_corr_part(const float* __restrict__ x, const float* __restrict__ y,
                            float* __restrict__ acc, int n)
{
  int i = blockIdx.x * blockDim.x + threadIdx.x;
  float a = 0.f, b = 0.f;
  if (i < n) { a = x[i]; b = y[i]; }
  float aa = a * a, bb = b * b, ab = a * b;
  #pragma unroll
  for (int off = 16; off; off >>= 1) {
    a  += __shfl_down(a,  off, 32);  b  += __shfl_down(b,  off, 32);
    aa += __shfl_down(aa, off, 32);  bb += __shfl_down(bb, off, 32);
    ab += __shfl_down(ab, off, 32);
  }
  if ((threadIdx.x & 31) == 0) {
    atomicAdd(acc + 0, a);  atomicAdd(acc + 1, b);
    atomicAdd(acc + 2, aa); atomicAdd(acc + 3, bb); atomicAdd(acc + 4, ab);
  }
}

__global__ void k_corr_final(const float* __restrict__ acc, float* __restrict__ out) {
  if (threadIdx.x != 0 || blockIdx.x != 0) return;
  const float inv = 1.f / (float)N_ENTC;
  float ex = acc[0] * inv, ey = acc[1] * inv;
  float v1 = acc[2] * inv - ex * ex;
  float v2 = acc[3] * inv - ey * ey;
  float cv = acc[4] * inv - ex * ey;
  out[0] = fabsf(cv) / (sqrtf(v1) * sqrtf(v2));
}

// ---------------- ConvE ----------------
__global__ void k_build_img(const int* __restrict__ h_id, const int* __restrict__ r_id,
                            const float* __restrict__ ent_out, const float* __restrict__ prel,
                            float* __restrict__ img)
{
  int idx = blockIdx.x * blockDim.x + threadIdx.x;       // BSZ*256
  if (idx >= BSZ * 256) return;
  int b = idx >> 8, i = idx & 255;
  int row = i >> 4, col = i & 15;
  float v;
  if (row < 8) v = ent_out[(size_t)h_id[b] * HDIM + row * 16 + col];
  else         v = prel  [(size_t)r_id[b] * HDIM + (row - 8) * 16 + col];
  img[idx] = v;
}

__global__ __launch_bounds__(256) void k_conv(const float* __restrict__ img,
                                              const float* __restrict__ cw,
                                              const float* __restrict__ cb,
                                              unsigned short* __restrict__ outbf)
{
  __shared__ float w[CCH * 49];
  __shared__ float x[256];
  int b = blockIdx.x;
  for (int i = threadIdx.x; i < CCH * 49; i += 256) w[i] = cw[i];
  x[threadIdx.x] = img[(size_t)b * 256 + threadIdx.x];
  __syncthreads();
  for (int o = threadIdx.x; o < FLATD; o += 256) {
    int c = o / 100, rem = o % 100, y = rem / 10, xx = rem % 10;
    float acc = cb[c];
    const float* wp = w + c * 49;
    #pragma unroll
    for (int ky = 0; ky < 7; ++ky)
      #pragma unroll
      for (int kx = 0; kx < 7; ++kx)
        acc += x[(y + ky) * 16 + (xx + kx)] * wp[ky * 7 + kx];
    outbf[(size_t)b * FLATD + o] = f2bf(fmaxf(acc, 0.f));
  }
}

// ---------------- launcher ----------------
extern "C" void kernel_launch(void* const* d_in, const int* in_sizes, int n_in,
                              void* d_out, int out_size, void* d_ws, size_t ws_size,
                              hipStream_t stream)
{
  (void)in_sizes; (void)n_in; (void)out_size; (void)ws_size;
  const int*   h_id    = (const int*)  d_in[0];
  const int*   r_id    = (const int*)  d_in[1];
  const int*   src     = (const int*)  d_in[2];
  const int*   dst     = (const int*)  d_in[3];
  const int*   rel_id  = (const int*)  d_in[4];
  const float* ent_emb = (const float*)d_in[5];
  const float* rel0    = (const float*)d_in[6];
  const float* neigh_w = (const float*)d_in[7];
  const float* S_w = (const float*)d_in[8];  const float* S_b = (const float*)d_in[9];
  const float* L_w = (const float*)d_in[10]; const float* L_b = (const float*)d_in[11];
  const float* prel    = (const float*)d_in[12];
  const float* phi_w0 = (const float*)d_in[13]; const float* phi_b0 = (const float*)d_in[14];
  const float* phi_w1 = (const float*)d_in[15]; const float* phi_b1 = (const float*)d_in[16];
  const float* phi_w2 = (const float*)d_in[17]; const float* phi_b2 = (const float*)d_in[18];
  const float* psi_w0 = (const float*)d_in[19]; const float* psi_b0 = (const float*)d_in[20];
  const float* psi_w1 = (const float*)d_in[21]; const float* psi_b1 = (const float*)d_in[22];
  const float* psi_w2 = (const float*)d_in[23]; const float* psi_b2 = (const float*)d_in[24];
  const float* conv_w = (const float*)d_in[25]; const float* conv_b = (const float*)d_in[26];
  const float* fc_w   = (const float*)d_in[27]; const float* fc_b   = (const float*)d_in[28];
  const float* score_b= (const float*)d_in[29];

  float* out_score = (float*)d_out;                       // 1024 x 50000
  float* out_corr  = out_score + (size_t)BSZ * N_ENTC;    // scalar

  // ---- workspace bump allocator (64KB tail pad per buffer for unguarded GEMM tiles) ----
  char* ws = (char*)d_ws; size_t off = 0;
  auto alloc = [&](size_t bytes) -> void* {
    void* p = ws + off; off = (off + bytes + 65536 + 255) & ~(size_t)255; return p;
  };
  unsigned short* ent_bf   = (unsigned short*)alloc((size_t)N_ENTC * HDIM * 2);
  unsigned short* wS_bf    = (unsigned short*)alloc(HDIM * HDIM * 2);
  unsigned short* wL_bf    = (unsigned short*)alloc(HDIM * HDIM * 2);
  unsigned short* wN_bf    = (unsigned short*)alloc(HDIM * HDIM * 2);
  unsigned short* wphi0_bf = (unsigned short*)alloc(HDIM * MLPP * 2);   // padded 128x224
  unsigned short* wphi1_bf = (unsigned short*)alloc(MLPP * MLPP * 2);   // padded 224x224
  unsigned short* wpsi0_bf = (unsigned short*)alloc(HDIM * MLPP * 2);
  unsigned short* wpsi1_bf = (unsigned short*)alloc(MLPP * MLPP * 2);
  unsigned short* wfc_bf   = (unsigned short*)alloc((size_t)FLATD * HDIM * 2);
  float* pb0 = (float*)alloc(MLPP * 4);   // padded biases (phi_b0, phi_b1, psi_b0, psi_b1)
  float* pb1 = (float*)alloc(MLPP * 4);
  float* pb2 = (float*)alloc(MLPP * 4);
  float* pb3 = (float*)alloc(MLPP * 4);
  float* common   = (float*)alloc((size_t)N_ENTC * HDIM * 4);
  float* privb    = (float*)alloc((size_t)N_ENTC * HDIM * 4);
  float* score_e  = (float*)alloc((size_t)EDGESC * 4);
  unsigned* m_node= (unsigned*)alloc((size_t)N_ENTC * 4);
  float* z_node   = (float*)alloc((size_t)N_ENTC * 4);
  unsigned* hist  = (unsigned*)alloc((size_t)N_ENTC * NBIN * 4);
  int* thr        = (int*)alloc((size_t)N_ENTC * 4);
  unsigned* bud   = (unsigned*)alloc((size_t)N_ENTC * 4);
  unsigned* cnt   = (unsigned*)alloc((size_t)N_ENTC * 4);
  float* neigh    = (float*)alloc((size_t)N_ENTC * HDIM * 4);
  unsigned short* neigh_bf = (unsigned short*)alloc((size_t)N_ENTC * HDIM * 2);
  float* c1       = (float*)alloc((size_t)N_ENTC * HDIM * 4);
  float* c2       = (float*)alloc((size_t)N_ENTC * HDIM * 4);
  unsigned short* c_bf = (unsigned short*)alloc((size_t)N_ENTC * HDIM * 2);
  float* ent_out  = (float*)alloc((size_t)N_ENTC * HDIM * 4);
  unsigned short* entout_bf = (unsigned short*)alloc((size_t)N_ENTC * HDIM * 2);
  float* a1       = (float*)alloc((size_t)N_ENTC * MLPP * 4);           // stride 224
  unsigned short* a1_bf = (unsigned short*)alloc((size_t)N_ENTC * MLPP * 2);
  float* out1     = (float*)alloc((size_t)N_ENTC * 4);
  float* out2     = (float*)alloc((size_t)N_ENTC * 4);
  float* acc5     = (float*)alloc(64);
  float* ximg     = (float*)alloc((size_t)BSZ * 256 * 4);
  unsigned short* conv_bf = (unsigned short*)alloc((size_t)BSZ * FLATD * 2);
  float* fc_out   = (float*)alloc((size_t)BSZ * HDIM * 4);
  unsigned short* fc_bf = (unsigned short*)alloc((size_t)BSZ * HDIM * 2);
  float* a2 = a1;  // GEMM2 reads a1_bf, so its f32 output can alias a1

  auto cvt = [&](const float* in, unsigned short* o, size_t n) {
    k_cvt_bf<<<CDIV(n, (size_t)256), 256, 0, stream>>>(in, o, n);
  };
  auto cvtpad = [&](const float* in, unsigned short* o, int r, int c, int rp, int cp) {
    k_cvt_pad<<<CDIV(rp * cp, 256), 256, 0, stream>>>(in, o, r, c, rp, cp);
  };
  auto fill = [&](void* p, unsigned v, size_t n) {
    k_fill_u32<<<CDIV(n, (size_t)256), 256, 0, stream>>>((unsigned*)p, v, n);
  };
  auto gemm = [&](const unsigned short* A, const unsigned short* B, const float* bias,
                  float* C, int M, int N, int K, int BT, int act) {
    dim3 g(CDIV(N, 128), CDIV(M, 64));
    k_gemm_bf16<<<g, 256, 0, stream>>>(A, B, bias, C, M, N, K, BT, act);
  };

  // ---- weight / embedding conversions ----
  cvt(ent_emb, ent_bf, (size_t)N_ENTC * HDIM);
  cvt(S_w, wS_bf, HDIM * HDIM);         cvt(L_w, wL_bf, HDIM * HDIM);
  cvt(neigh_w, wN_bf, HDIM * HDIM);
  cvtpad(phi_w0, wphi0_bf, HDIM, MLPD, HDIM, MLPP);
  cvtpad(phi_w1, wphi1_bf, MLPD, MLPD, MLPP, MLPP);
  cvtpad(psi_w0, wpsi0_bf, HDIM, MLPD, HDIM, MLPP);
  cvtpad(psi_w1, wpsi1_bf, MLPD, MLPD, MLPP, MLPP);
  cvt(fc_w, wfc_bf, (size_t)FLATD * HDIM);
  k_pad_vec<<<1, MLPP, 0, stream>>>(phi_b0, pb0, MLPD, MLPP);
  k_pad_vec<<<1, MLPP, 0, stream>>>(phi_b1, pb1, MLPD, MLPP);
  k_pad_vec<<<1, MLPP, 0, stream>>>(psi_b0, pb2, MLPD, MLPP);
  k_pad_vec<<<1, MLPP, 0, stream>>>(psi_b1, pb3, MLPD, MLPP);

  // ---- common / private projections (WMMA) ----
  gemm(ent_bf, wS_bf, S_b, common, N_ENTC, HDIM, HDIM, 0, 0);
  gemm(ent_bf, wL_bf, L_b, privb,  N_ENTC, HDIM, HDIM, 0, 0);

  // ---- CompGCN layer (edge softmax + top-k + aggregate + tanh GEMM) ----
  auto run_layer = [&](const float* nodeEmb, float* cOut) {
    fill(m_node, 0x007FFFFFu, N_ENTC);                    // ordered(-inf)
    fill(z_node, 0u, N_ENTC);
    fill(hist,   0u, (size_t)N_ENTC * NBIN);
    fill(neigh,  0u, (size_t)N_ENTC * HDIM);
    k_edge_score<<<EDGESC / 8, 256, 0, stream>>>(nodeEmb, rel0, src, dst, rel_id,
                                                 score_e, m_node, EDGESC);
    k_edge_expz<<<CDIV(EDGESC, 256), 256, 0, stream>>>(score_e, m_node, dst,
                                                       z_node, hist, EDGESC);
    k_node_select<<<CDIV(N_ENTC, 256), 256, 0, stream>>>(hist, thr, bud, cnt, N_ENTC);
    k_edge_agg<<<EDGESC / 8, 256, 0, stream>>>(nodeEmb, rel0, src, dst, rel_id,
                                               score_e, m_node, z_node, thr, bud, cnt,
                                               neigh, EDGESC);
    cvt(neigh, neigh_bf, (size_t)N_ENTC * HDIM);
    gemm(neigh_bf, wN_bf, nullptr, cOut, N_ENTC, HDIM, HDIM, 0, 2);   // tanh
  };
  run_layer(common, c1);
  run_layer(privb,  c2);

  // ---- ent_out = ent_emb + c1 + c2 (+ bf16 copy for scoring GEMM) ----
  {
    size_t n = (size_t)N_ENTC * HDIM;
    k_add3<<<CDIV(n, (size_t)256), 256, 0, stream>>>(ent_emb, c1, c2, ent_out, entout_bf, n);
  }

  // ---- phi / psi MLPs + correlation (padded width 224; pad lanes compute 0) ----
  auto run_mlp = [&](const float* cin, const unsigned short* w0bf, const float* b0p,
                     const unsigned short* w1bf, const float* b1p,
                     const float* w2, const float* b2, float* ov) {
    cvt(cin, c_bf, (size_t)N_ENTC * HDIM);
    gemm(c_bf, w0bf, b0p, a1, N_ENTC, MLPP, HDIM, 0, 1);              // relu
    cvt(a1, a1_bf, (size_t)N_ENTC * MLPP);
    gemm(a1_bf, w1bf, b1p, a2, N_ENTC, MLPP, MLPP, 0, 1);             // relu
    k_rowdot<<<CDIV((size_t)N_ENTC * 32, (size_t)256), 256, 0, stream>>>(
        a2, w2, b2, ov, N_ENTC, MLPD, MLPP);
  };
  run_mlp(c1, wphi0_bf, pb0, wphi1_bf, pb1, phi_w2, phi_b2, out1);
  run_mlp(c2, wpsi0_bf, pb2, wpsi1_bf, pb3, psi_w2, psi_b2, out2);
  fill(acc5, 0u, 8);
  k_corr_part<<<CDIV(N_ENTC, 256), 256, 0, stream>>>(out1, out2, acc5, N_ENTC);
  k_corr_final<<<1, 32, 0, stream>>>(acc5, out_corr);

  // ---- ConvE ----
  k_build_img<<<CDIV(BSZ * 256, 256), 256, 0, stream>>>(h_id, r_id, ent_out, prel, ximg);
  k_conv<<<BSZ, 256, 0, stream>>>(ximg, conv_w, conv_b, conv_bf);
  gemm(conv_bf, wfc_bf, fc_b, fc_out, BSZ, HDIM, FLATD, 0, 1);        // relu
  cvt(fc_out, fc_bf, (size_t)BSZ * HDIM);
  // score = sigmoid(x @ ent_out^T + score_b)  (BT GEMM against bf16 ent_out)
  gemm(fc_bf, entout_bf, score_b, out_score, BSZ, N_ENTC, HDIM, 1, 3);
}
